// NeuronArchitecture_images_11922829214361
// MI455X (gfx1250) — matrix operand order repaired and checked by reading, not compiled, not run on hardware
//
#include <hip/hip_runtime.h>
#include <hip/hip_bf16.h>

typedef __attribute__((ext_vector_type(16))) _Float16 v16h;
typedef __attribute__((ext_vector_type(8)))  _Float16 v8h;
typedef __attribute__((ext_vector_type(8)))  float    v8f;

#define D0_   512
#define DH_   128
#define DF_   64
#define NB_   64
#define LL_   16
#define NSEG_ 1024
#define NTOT_ (NB_*LL_*256)   /* 262144 */
#define EPSBN 1e-5f

// ---------------------------------------------------------------------------
// small utility kernels
// ---------------------------------------------------------------------------
__global__ void seg_kernel(const int* __restrict__ act, const int* __restrict__ bid,
                           const int* __restrict__ Lp, int* __restrict__ seg, int n) {
  int i = blockIdx.x * blockDim.x + threadIdx.x;
  if (i < n) seg[i] = act[i] + bid[i] * Lp[0];
}

__global__ void zero_kernel(float* __restrict__ p, int n) {
  int i = blockIdx.x * blockDim.x + threadIdx.x;
  if (i < n) p[i] = 0.0f;
}

// Pack a row-major f32 weight [K x Nout] into wave32 WMMA B-fragment order:
// dst[((kt*ntiles + nt)*32 + lane)*16 + e] = (half)src[(kt*32 + (lane>>4)*8 + (e&7) + (e>>3)*16)*Nout + nt*16 + (lane&15)]
__global__ void pack_w_kernel(const float* __restrict__ src, _Float16* __restrict__ dst,
                              int K, int Nout) {
  int idx = blockIdx.x * blockDim.x + threadIdx.x;
  int total = K * Nout;
  if (idx >= total) return;
  int e    = idx & 15;
  int lane = (idx >> 4) & 31;
  int tile = idx >> 9;
  int ntiles = Nout >> 4;
  int nt = tile % ntiles;
  int kt = tile / ntiles;
  int g  = lane >> 4;
  int n  = nt * 16 + (lane & 15);
  int k  = kt * 32 + g * 8 + (e & 7) + (e >> 3) * 16;
  dst[idx] = (_Float16)src[(size_t)k * Nout + n];
}

// segment scatter-sum of raw rows: s[seg[node]][:] += H[node][:]
template<int KIN>
__global__ void segsum_kernel(const float* __restrict__ H, const int* __restrict__ seg,
                              float* __restrict__ s, int nNodes) {
  int i = blockIdx.x * blockDim.x + threadIdx.x;
  int total = nNodes * (KIN / 4);
  if (i >= total) return;
  int node = i / (KIN / 4);
  int c4   = i % (KIN / 4);
  float4 v = ((const float4*)H)[i];
  int sgv = seg[node];
  float* base = s + (size_t)sgv * KIN + c4 * 4;
  unsafeAtomicAdd(base + 0, v.x);
  unsafeAtomicAdd(base + 1, v.y);
  unsafeAtomicAdd(base + 2, v.z);
  unsafeAtomicAdd(base + 3, v.w);
}

// BatchNorm finalize (+ optional residual), h = [res? h:0] + gamma*(z-m)*rsqrt(v+eps)+beta
template<bool RES>
__global__ void bn_residual_kernel(const float* __restrict__ z,
                                   const float* __restrict__ gSum, const float* __restrict__ gSq,
                                   const float* __restrict__ gamma, const float* __restrict__ beta,
                                   float* __restrict__ h, int n4) {
  int i = blockIdx.x * blockDim.x + threadIdx.x;
  if (i >= n4) return;
  int c = (i & (DH_ / 4 - 1)) * 4;
  const float invN = 1.0f / (float)NTOT_;
  float4 zv = ((const float4*)z)[i];
  float o[4];
  float zz[4] = {zv.x, zv.y, zv.z, zv.w};
#pragma unroll
  for (int j = 0; j < 4; ++j) {
    float m   = gSum[c + j] * invN;
    float var = gSq[c + j] * invN - m * m;
    float is  = rsqrtf(var + EPSBN);
    o[j] = (zz[j] - m) * is * gamma[c + j] + beta[c + j];
  }
  if (RES) {
    float4 hv = ((const float4*)h)[i];
    o[0] += hv.x; o[1] += hv.y; o[2] += hv.z; o[3] += hv.w;
  }
  ((float4*)h)[i] = make_float4(o[0], o[1], o[2], o[3]);
}

// pooled[b] = sum_l r[b*L+l], out[b] = sigmoid(pooled . outW + outB)
__global__ void final_kernel(const float* __restrict__ r, const float* __restrict__ outW,
                             const float* __restrict__ outB, float* __restrict__ out) {
  __shared__ float red[DF_];
  int b = blockIdx.x;
  int f = threadIdx.x;
  float p = 0.0f;
#pragma unroll
  for (int l = 0; l < LL_; ++l) p += r[((size_t)b * LL_ + l) * DF_ + f];
  red[f] = p * outW[f];
  __syncthreads();
  for (int st = DF_ / 2; st > 0; st >>= 1) {
    if (f < st) red[f] += red[f + st];
    __syncthreads();
  }
  if (f == 0) out[b] = 1.0f / (1.0f + expf(-(red[0] + outB[0])));
}

// ---------------------------------------------------------------------------
// fused 2-layer MLP (the workhorse): Z = relu(H@W1+b1)@W2+b2  via f16 WMMA
// ---------------------------------------------------------------------------
enum { MODE_STORE = 0, MODE_PHI_BN = 1, MODE_POOL = 2 };

// A fragment (16x32 f16) from an f32 row-major matrix; lane row = lane&15.
__device__ inline v16h load_a_f32(const float* __restrict__ rowptr, int k0, int g) {
  const float4* p0 = (const float4*)(rowptr + k0 + g * 8);
  const float4* p1 = (const float4*)(rowptr + k0 + 16 + g * 8);
  float4 u0 = p0[0], u1 = p0[1], u2 = p1[0], u3 = p1[1];
  v16h a;
  a[0] = (_Float16)u0.x; a[1] = (_Float16)u0.y; a[2]  = (_Float16)u0.z; a[3]  = (_Float16)u0.w;
  a[4] = (_Float16)u1.x; a[5] = (_Float16)u1.y; a[6]  = (_Float16)u1.z; a[7]  = (_Float16)u1.w;
  a[8] = (_Float16)u2.x; a[9] = (_Float16)u2.y; a[10] = (_Float16)u2.z; a[11] = (_Float16)u2.w;
  a[12] = (_Float16)u3.x; a[13] = (_Float16)u3.y; a[14] = (_Float16)u3.z; a[15] = (_Float16)u3.w;
  return a;
}

// A fragment from LDS-resident f16 row (pitch handled by caller)
__device__ inline v16h load_a_lds(const _Float16* base, int k0, int g) {
  v8h lo = *(const v8h*)(base + k0 + g * 8);
  v8h hi = *(const v8h*)(base + k0 + 16 + g * 8);
  v16h a;
#pragma unroll
  for (int e = 0; e < 8; ++e) { a[e] = lo[e]; a[8 + e] = hi[e]; }
  return a;
}

#define LDS_PITCH 136  /* halves; 272B keeps 16B alignment, avoids bank conflicts */

template<int KIN, int NOUT, int MODE>
__global__ __launch_bounds__(256) void fused_mlp_kernel(
    const float* __restrict__ H,
    const _Float16* __restrict__ W1p, const float* __restrict__ b1,
    const _Float16* __restrict__ W2p, const float* __restrict__ b2,
    const int* __restrict__ seg, const float* __restrict__ rs,
    float* __restrict__ out, float* __restrict__ gSum, float* __restrict__ gSq) {
  constexpr int T2 = NOUT / 16;
  __shared__ _Float16 Tl[128 * LDS_PITCH];
  __shared__ float lsum[DH_], lsq[DH_];

  const int tid  = threadIdx.x;
  const int wave = tid >> 5;
  const int lane = tid & 31;
  const int g    = lane >> 4;
  const int nn   = lane & 15;
  const int rowBase = blockIdx.x * 128 + wave * 16;

  if constexpr (MODE == MODE_PHI_BN) {
    if (tid < DH_) { lsum[tid] = 0.0f; lsq[tid] = 0.0f; }
  }
  __syncthreads();

  // ---- stage 1: T = relu(H @ W1 + b1), 16 rows x 128 cols per wave ----
  v8f acc1[8];
#pragma unroll
  for (int t = 0; t < 8; ++t)
#pragma unroll
    for (int r = 0; r < 8; ++r) acc1[t][r] = 0.0f;

  const float* Hrow = H + (size_t)(rowBase + nn) * KIN;
  for (int kt = 0; kt < KIN / 32; ++kt) {
    v16h a = load_a_f32(Hrow, kt * 32, g);
#pragma unroll
    for (int t = 0; t < 8; ++t) {
      v16h b = *(const v16h*)(W1p + ((size_t)(kt * 8 + t) * 32 + lane) * 16);
      acc1[t] = __builtin_amdgcn_wmma_f32_16x16x32_f16(
          false, a, false, b, (short)0, acc1[t], false, false);
    }
  }

  // bias + relu, spill to LDS as f16 (D layout: row = g*8+r, col = t*16+nn)
  _Float16* Tw = &Tl[(size_t)wave * 16 * LDS_PITCH];
#pragma unroll
  for (int t = 0; t < 8; ++t) {
    float bb = b1[t * 16 + nn];
#pragma unroll
    for (int r = 0; r < 8; ++r) {
      float v = acc1[t][r] + bb;
      v = v > 0.0f ? v : 0.0f;
      Tw[(g * 8 + r) * LDS_PITCH + t * 16 + nn] = (_Float16)v;
    }
  }
  __syncthreads();

  // ---- stage 2: Z = T @ W2 + b2 ----
  v8f acc2[T2];
#pragma unroll
  for (int t = 0; t < T2; ++t)
#pragma unroll
    for (int r = 0; r < 8; ++r) acc2[t][r] = 0.0f;

  const _Float16* Trow = &Tl[((size_t)wave * 16 + nn) * LDS_PITCH];
#pragma unroll
  for (int kt = 0; kt < DH_ / 32; ++kt) {
    v16h a = load_a_lds(Trow, kt * 32, g);
#pragma unroll
    for (int t = 0; t < T2; ++t) {
      v16h b = *(const v16h*)(W2p + ((size_t)(kt * T2 + t) * 32 + lane) * 16);
      acc2[t] = __builtin_amdgcn_wmma_f32_16x16x32_f16(
          false, a, false, b, (short)0, acc2[t], false, false);
    }
  }

  // ---- epilogue ----
  const int rowG0 = rowBase + g * 8;
  int sg[8];
  if constexpr (MODE == MODE_PHI_BN || MODE == MODE_POOL) {
#pragma unroll
    for (int r = 0; r < 8; ++r) sg[r] = seg[rowG0 + r];
  }

#pragma unroll
  for (int t = 0; t < T2; ++t) {
    int col = t * 16 + nn;
    float bb = b2[col];
    float psum = 0.0f, psq = 0.0f;
#pragma unroll
    for (int r = 0; r < 8; ++r) {
      float v = acc2[t][r] + bb;
      if constexpr (MODE == MODE_STORE) {
        out[(size_t)(rowG0 + r) * NOUT + col] = v;
      } else if constexpr (MODE == MODE_POOL) {
        unsafeAtomicAdd(&out[(size_t)sg[r] * DH_ + col], v);
      } else {  // MODE_PHI_BN : add rho gather, store z, accumulate BN stats
        v += rs[(size_t)sg[r] * DH_ + col];
        out[(size_t)(rowG0 + r) * DH_ + col] = v;
        psum += v;
        psq  += v * v;
      }
    }
    if constexpr (MODE == MODE_PHI_BN) {
      atomicAdd(&lsum[col], psum);
      atomicAdd(&lsq[col], psq);
    }
  }

  if constexpr (MODE == MODE_PHI_BN) {
    __syncthreads();
    if (tid < DH_) {
      unsafeAtomicAdd(&gSum[tid], lsum[tid]);
      unsafeAtomicAdd(&gSq[tid],  lsq[tid]);
    }
  }
}

// ---------------------------------------------------------------------------
// host side
// ---------------------------------------------------------------------------
struct PP {
  const float *x;
  const float *phiW1[4], *phib1[4], *phiW2[4], *phib2[4];  // [3] = pool
  const float *rhoW1[4], *rhob1[4], *rhoW2[4], *rhob2[4];
  const float *gamma[3], *beta[3];
  const float *outW, *outB;
  const int *act, *bid, *Lp;
};

static void map_inputs(void* const* d_in, const int* in_sizes, PP& P) {
  // insertion order => x (N*512 elements) is first; JAX-sorted pytree => activation_idx first.
  bool insertion = in_sizes[0] > 1000000;
  int i = 0;
  auto F = [&]() -> const float* { return (const float*)d_in[i++]; };
  auto I = [&]() -> const int*   { return (const int*)d_in[i++]; };
  if (insertion) {
    P.x = F();
    for (int l = 0; l < 3; ++l) {
      P.phiW1[l] = F(); P.phib1[l] = F(); P.phiW2[l] = F(); P.phib2[l] = F();
      P.rhoW1[l] = F(); P.rhob1[l] = F(); P.rhoW2[l] = F(); P.rhob2[l] = F();
    }
    for (int l = 0; l < 3; ++l) { P.gamma[l] = F(); P.beta[l] = F(); }
    P.phiW1[3] = F(); P.phib1[3] = F(); P.phiW2[3] = F(); P.phib2[3] = F();
    P.rhoW1[3] = F(); P.rhob1[3] = F(); P.rhoW2[3] = F(); P.rhob2[3] = F();
    P.outW = F(); P.outB = F();
    P.act = I(); P.bid = I(); P.Lp = I();
  } else {
    P.act = I(); P.bid = I(); P.Lp = I();
    for (int l = 0; l < 3; ++l) { P.beta[l] = F(); P.gamma[l] = F(); }
    for (int l = 0; l < 3; ++l) {
      P.phiW1[l] = F(); P.phiW2[l] = F(); P.phib1[l] = F(); P.phib2[l] = F();
      P.rhoW1[l] = F(); P.rhoW2[l] = F(); P.rhob1[l] = F(); P.rhob2[l] = F();
    }
    P.outW = F(); P.outB = F();
    P.phiW1[3] = F(); P.phiW2[3] = F(); P.phib1[3] = F(); P.phib2[3] = F();
    P.rhoW1[3] = F(); P.rhoW2[3] = F(); P.rhob1[3] = F(); P.rhob2[3] = F();
    P.x = F();
  }
}

static inline size_t alignup(size_t v) { return (v + 255) & ~(size_t)255; }

extern "C" void kernel_launch(void* const* d_in, const int* in_sizes, int n_in,
                              void* d_out, int out_size, void* d_ws, size_t ws_size,
                              hipStream_t stream) {
  (void)n_in; (void)out_size; (void)ws_size;
  PP P;
  map_inputs(d_in, in_sizes, P);

  // --- workspace carve-out ---
  char* w = (char*)d_ws;
  auto grab = [&](size_t bytes) { char* r = w; w += alignup(bytes); return r; };
  int*   seg  = (int*)grab((size_t)NTOT_ * 4);
  float* z    = (float*)grab((size_t)NTOT_ * DH_ * 4);
  float* h    = (float*)grab((size_t)NTOT_ * DH_ * 4);
  float* s    = (float*)grab((size_t)NSEG_ * D0_ * 4);     // reused, layer0 is biggest
  float* rsb  = (float*)grab((size_t)NSEG_ * DH_ * 4);
  float* rp   = (float*)grab((size_t)NSEG_ * DF_ * 4);
  float* gS   = (float*)grab(DH_ * 4);
  float* gQ   = (float*)grab(DH_ * 4);
  _Float16* phiW1p[4], *phiW2p[4], *rhoW1p[4], *rhoW2p[4];
  for (int l = 0; l < 4; ++l) {
    int k1 = (l == 0) ? D0_ : DH_;
    phiW1p[l] = (_Float16*)grab((size_t)k1 * DH_ * 2);
    phiW2p[l] = (_Float16*)grab((size_t)DH_ * DH_ * 2);
    rhoW1p[l] = (_Float16*)grab((size_t)k1 * DH_ * 2);
    int n2 = (l == 3) ? DF_ : DH_;
    rhoW2p[l] = (_Float16*)grab((size_t)DH_ * n2 * 2);
  }

  const int TB = 256;
  auto blocks = [](long n, int tb) { return (int)((n + tb - 1) / tb); };

  // seg ids
  seg_kernel<<<blocks(NTOT_, TB), TB, 0, stream>>>(P.act, P.bid, P.Lp, seg, NTOT_);

  // pack weights into WMMA-B fragment layout (f16)
  for (int l = 0; l < 4; ++l) {
    int k1 = (l == 0) ? D0_ : DH_;
    int n2 = (l == 3) ? DF_ : DH_;
    pack_w_kernel<<<blocks((long)k1 * DH_, TB), TB, 0, stream>>>(P.phiW1[l], phiW1p[l], k1, DH_);
    pack_w_kernel<<<blocks((long)DH_ * DH_, TB), TB, 0, stream>>>(P.phiW2[l], phiW2p[l], DH_, DH_);
    pack_w_kernel<<<blocks((long)k1 * DH_, TB), TB, 0, stream>>>(P.rhoW1[l], rhoW1p[l], k1, DH_);
    pack_w_kernel<<<blocks((long)DH_ * n2, TB), TB, 0, stream>>>(P.rhoW2[l], rhoW2p[l], DH_, n2);
  }

  // --- 3 DeepSets layers ---
  for (int l = 0; l < 3; ++l) {
    const float* Hin = (l == 0) ? P.x : h;
    const int KIN = (l == 0) ? D0_ : DH_;

    zero_kernel<<<blocks(NSEG_ * KIN, TB), TB, 0, stream>>>(s, NSEG_ * KIN);
    if (l == 0)
      segsum_kernel<D0_><<<blocks((long)NTOT_ * (D0_ / 4), TB), TB, 0, stream>>>(Hin, seg, s, NTOT_);
    else
      segsum_kernel<DH_><<<blocks((long)NTOT_ * (DH_ / 4), TB), TB, 0, stream>>>(Hin, seg, s, NTOT_);

    // rho MLP on the 1024 segment rows
    if (l == 0)
      fused_mlp_kernel<D0_, DH_, MODE_STORE><<<NSEG_ / 128, TB, 0, stream>>>(
          s, rhoW1p[l], P.rhob1[l], rhoW2p[l], P.rhob2[l], nullptr, nullptr, rsb, nullptr, nullptr);
    else
      fused_mlp_kernel<DH_, DH_, MODE_STORE><<<NSEG_ / 128, TB, 0, stream>>>(
          s, rhoW1p[l], P.rhob1[l], rhoW2p[l], P.rhob2[l], nullptr, nullptr, rsb, nullptr, nullptr);

    // phi MLP on all nodes + rho gather + BN stats
    zero_kernel<<<1, TB, 0, stream>>>(gS, DH_);
    zero_kernel<<<1, TB, 0, stream>>>(gQ, DH_);
    if (l == 0)
      fused_mlp_kernel<D0_, DH_, MODE_PHI_BN><<<NTOT_ / 128, TB, 0, stream>>>(
          Hin, phiW1p[l], P.phib1[l], phiW2p[l], P.phib2[l], seg, rsb, z, gS, gQ);
    else
      fused_mlp_kernel<DH_, DH_, MODE_PHI_BN><<<NTOT_ / 128, TB, 0, stream>>>(
          Hin, phiW1p[l], P.phib1[l], phiW2p[l], P.phib2[l], seg, rsb, z, gS, gQ);

    // BN finalize (+ residual for layers 1,2)
    int n4 = NTOT_ * DH_ / 4;
    if (l == 0)
      bn_residual_kernel<false><<<blocks(n4, TB), TB, 0, stream>>>(z, gS, gQ, P.gamma[l], P.beta[l], h, n4);
    else
      bn_residual_kernel<true><<<blocks(n4, TB), TB, 0, stream>>>(z, gS, gQ, P.gamma[l], P.beta[l], h, n4);
  }

  // --- invariant pooling head ---
  zero_kernel<<<blocks(NSEG_ * DH_, TB), TB, 0, stream>>>(s, NSEG_ * DH_);
  fused_mlp_kernel<DH_, DH_, MODE_POOL><<<NTOT_ / 128, TB, 0, stream>>>(
      h, phiW1p[3], P.phib1[3], phiW2p[3], P.phib2[3], seg, nullptr, s, nullptr, nullptr);
  fused_mlp_kernel<DH_, DF_, MODE_STORE><<<NSEG_ / 128, TB, 0, stream>>>(
      s, rhoW1p[3], P.rhob1[3], rhoW2p[3], P.rhob2[3], nullptr, nullptr, rp, nullptr, nullptr);
  final_kernel<<<NB_, DF_, 0, stream>>>(rp, P.outW, P.outB, (float*)d_out);
}
